// SimpleAttention_74002286510487
// MI455X (gfx1250) — matrix operand order repaired and checked by reading, not compiled, not run on hardware
//
#include <hip/hip_runtime.h>
#include <hip/hip_bf16.h>

typedef __attribute__((ext_vector_type(16))) __bf16 v16bf;
typedef __attribute__((ext_vector_type(8)))  __bf16 v8bf;
typedef __attribute__((ext_vector_type(8)))  float  v8f;
typedef unsigned int u32;
typedef __attribute__((ext_vector_type(4))) unsigned int u32x4;
typedef __attribute__((ext_vector_type(4))) int          i32x4;
typedef __attribute__((ext_vector_type(8))) int          i32x8;

static constexpr float ATTN_SCALE = 0.125f;   // 64^-0.5

#define WMMA_BF16(a, b, c) \
  __builtin_amdgcn_wmma_f32_16x16x32_bf16(false, (a), false, (b), (short)0, (c), false, false)

#if defined(__has_builtin)
#if __has_builtin(__builtin_amdgcn_tensor_load_to_lds) && \
    __has_builtin(__builtin_amdgcn_s_wait_tensorcnt)
#define HAVE_TDM 1
#endif
#endif
#ifndef HAVE_TDM
#define HAVE_TDM 0
#endif

static __device__ __forceinline__ v16bf cat8(v8bf lo, v8bf hi) {
  return __builtin_shufflevector(lo, hi, 0,1,2,3,4,5,6,7,8,9,10,11,12,13,14,15);
}
static __device__ __forceinline__ v8f vzero() {
  v8f z;
#pragma unroll
  for (int i = 0; i < 8; ++i) z[i] = 0.0f;
  return z;
}

// 16-lane xor-butterfly reductions via DS_SWIZZLE_B32 (group-of-32 mode:
// offset = xor_mask<<10 | and_mask(0x1F)); stays inside each 16-lane half.
static __device__ __forceinline__ float swz_max16(float x) {
  x = fmaxf(x, __int_as_float(__builtin_amdgcn_ds_swizzle(__float_as_int(x), 0x041F)));
  x = fmaxf(x, __int_as_float(__builtin_amdgcn_ds_swizzle(__float_as_int(x), 0x081F)));
  x = fmaxf(x, __int_as_float(__builtin_amdgcn_ds_swizzle(__float_as_int(x), 0x101F)));
  x = fmaxf(x, __int_as_float(__builtin_amdgcn_ds_swizzle(__float_as_int(x), 0x201F)));
  return x;
}
static __device__ __forceinline__ float swz_add16(float x) {
  x += __int_as_float(__builtin_amdgcn_ds_swizzle(__float_as_int(x), 0x041F));
  x += __int_as_float(__builtin_amdgcn_ds_swizzle(__float_as_int(x), 0x081F));
  x += __int_as_float(__builtin_amdgcn_ds_swizzle(__float_as_int(x), 0x101F));
  x += __int_as_float(__builtin_amdgcn_ds_swizzle(__float_as_int(x), 0x201F));
  return x;
}

#if HAVE_TDM
// TDM 2-D tile load: global (row-major, bf16) -> LDS, with optional per-row
// LDS padding so the staged tile lands with our padded row stride.
// D# layout per CDNA5 ISA.
static __device__ __forceinline__ void tdm_load_2d(
    u32 lds_off, const void* gptr,
    u32 row_len,            // tensor_dim0 == tensor_dim0_stride (elements)
    u32 tile_w, u32 tile_h, // tile_dim0 / tile_dim1 (elements / rows)
    u32 pad_i, u32 pad_a,   // pad_interval code, pad_amount code
    bool pad_en) {
  unsigned long long ga = (unsigned long long)gptr;
  u32x4 g0;
  g0[0] = 1u;                                             // count=1, user mode
  g0[1] = lds_off;                                        // lds_addr
  g0[2] = (u32)ga;                                        // global_addr[31:0]
  g0[3] = ((u32)(ga >> 32) & 0x01FFFFFFu) | (2u << 30);   // addr[56:32] | type=2
  i32x8 g1;
  g1[0] = (int)((1u << 16) |                              // data_size=1 (2 bytes)
                (pad_en ? (1u << 20) : 0u) |              // pad_enable
                (pad_i << 22) | (pad_a << 25));           // pad_interval/amount
  g1[1] = (int)((row_len & 0xFFFFu) << 16);               // tensor_dim0[15:0]
  g1[2] = (int)(((row_len >> 16) & 0xFFFFu) |             // tensor_dim0[31:16]
                ((1u << 20 & 0xFFFFu) << 16));            // tensor_dim1[15:0] (big)
  g1[3] = (int)(((1u << 20) >> 16) |                      // tensor_dim1[31:16]
                (tile_w << 16));                          // tile_dim0
  g1[4] = (int)tile_h;                                    // tile_dim1 (tile_dim2=0)
  g1[5] = (int)row_len;                                   // tensor_dim0_stride[31:0]
  g1[6] = 0;                                              // stride0[47:32], stride1 lo
  g1[7] = 0;
  i32x4 z4 = {0, 0, 0, 0};
#if __clang_major__ >= 23
  i32x8 z8;
#pragma unroll
  for (int i = 0; i < 8; ++i) z8[i] = 0;
  __builtin_amdgcn_tensor_load_to_lds(g0, g1, z4, z4, z8, 0);
#else
  __builtin_amdgcn_tensor_load_to_lds(g0, g1, z4, z4, 0);
#endif
}
#endif

// ---------------------------------------------------------------------------
// f32 -> bf16 cast, 4 elements/thread
// ---------------------------------------------------------------------------
__global__ __launch_bounds__(256)
void cast_bf16_kernel(const float* __restrict__ in, __bf16* __restrict__ out, int n4) {
  int i = blockIdx.x * 256 + threadIdx.x;
  if (i < n4) {
    float4 v = ((const float4*)in)[i];
    union { __bf16 h[4]; uint2 u; } cv;
    cv.h[0] = (__bf16)v.x; cv.h[1] = (__bf16)v.y;
    cv.h[2] = (__bf16)v.z; cv.h[3] = (__bf16)v.w;
    ((uint2*)out)[i] = cv.u;
  }
}

// ---------------------------------------------------------------------------
// C[m,o] = sum_k A[m,k] * W[o,k] + bias[o]      (K fixed = 1024)
// BM = BN = 128, BK = 32, 8 waves; wave tile 64(M) x 32(N) = 4x2 WMMA tiles.
// A/B tiles staged to LDS by the Tensor Data Mover (padding applied in-flight).
// MODE 0: scatter to per-head bf16 Q (x scale) / K / V buffers [B,H,N,64]
// MODE 1: store f32 to outF [M, Otot]
// ---------------------------------------------------------------------------
template<int MODE>
__global__ __launch_bounds__(256)
void gemm_bias_kernel(const __bf16* __restrict__ A, const __bf16* __restrict__ W,
                      const float* __restrict__ bias,
                      float* __restrict__ outF,
                      __bf16* __restrict__ Qo, __bf16* __restrict__ Ko,
                      __bf16* __restrict__ Vo, int Otot) {
  __shared__ __align__(16) __bf16 As[128][48];   // 96B row stride (16B multiple)
  __shared__ __align__(16) __bf16 Bs[128][48];

  const int tid  = threadIdx.x;
  const int lane = tid & 31, wave = tid >> 5;
  const int wm = wave & 1, wn = wave >> 1;
  const int half = lane >> 4, l16 = lane & 15;
  const int rowBase = blockIdx.y * 128;
  const int colBase = blockIdx.x * 128;

  v8f acc[4][2];
#pragma unroll
  for (int i = 0; i < 4; ++i)
#pragma unroll
    for (int j = 0; j < 2; ++j) acc[i][j] = vzero();

  for (int k0 = 0; k0 < 1024; k0 += 32) {
    __syncthreads();
#if HAVE_TDM
    // 128x32 bf16 tile; LDS rows padded 16 DW data + 8 DW pad -> 48-elem stride
    if (wave == 0) {
      tdm_load_2d((u32)(uintptr_t)&As[0][0],
                  A + (size_t)rowBase * 1024 + k0,
                  1024, 32, 128, /*pad_i=*/3, /*pad_a=*/7, true);
      __builtin_amdgcn_s_wait_tensorcnt(0);
    } else if (wave == 1) {
      tdm_load_2d((u32)(uintptr_t)&Bs[0][0],
                  W + (size_t)colBase * 1024 + k0,
                  1024, 32, 128, /*pad_i=*/3, /*pad_a=*/7, true);
      __builtin_amdgcn_s_wait_tensorcnt(0);
    }
#else
    {
      const u32* Ag = (const u32*)A;
      const u32* Wg = (const u32*)W;
#pragma unroll
      for (int i = 0; i < 8; ++i) {
        int idx = tid + i * 256;            // 0..2047 -> 128 rows x 16 u32
        int r = idx >> 4, c = idx & 15;
        *(u32*)&As[r][c * 2] = Ag[(size_t)(rowBase + r) * 512 + (k0 >> 1) + c];
        *(u32*)&Bs[r][c * 2] = Wg[(size_t)(colBase + r) * 512 + (k0 >> 1) + c];
        if (k0 + 32 < 1024) {
          __builtin_prefetch(&Ag[(size_t)(rowBase + r) * 512 + ((k0 + 32) >> 1) + c], 0, 0);
          __builtin_prefetch(&Wg[(size_t)(colBase + r) * 512 + ((k0 + 32) >> 1) + c], 0, 0);
        }
      }
    }
#endif
    __syncthreads();

    // A fragments (16x32): elem e -> k = (e&7) + ((e>>3)<<4) + (half<<3)
    v16bf af[4];
#pragma unroll
    for (int mt = 0; mt < 4; ++mt) {
      int row = wm * 64 + mt * 16 + l16;
      af[mt] = cat8(*(const v8bf*)&As[row][half * 8],
                    *(const v8bf*)&As[row][16 + half * 8]);
    }
    // B fragments (32x16): lane = n, elem e -> k = (e&7) + ((e>>3)<<4) + (half<<3)
    v16bf bf2[2];
#pragma unroll
    for (int nt = 0; nt < 2; ++nt) {
      int col = wn * 32 + nt * 16 + l16;
      bf2[nt] = cat8(*(const v8bf*)&Bs[col][half * 8],
                     *(const v8bf*)&Bs[col][16 + half * 8]);
    }
#pragma unroll
    for (int mt = 0; mt < 4; ++mt)
#pragma unroll
      for (int nt = 0; nt < 2; ++nt)
        acc[mt][nt] = WMMA_BF16(af[mt], bf2[nt], acc[mt][nt]);
  }

  // Epilogue. C layout: m = r + 8*half, n = l16 (within each 16x16 tile)
#pragma unroll
  for (int mt = 0; mt < 4; ++mt) {
#pragma unroll
    for (int nt = 0; nt < 2; ++nt) {
      int go = colBase + wn * 32 + nt * 16 + l16;
      float bv = bias[go];
#pragma unroll
      for (int r = 0; r < 8; ++r) {
        int gm = rowBase + wm * 64 + mt * 16 + half * 8 + r;
        float v = acc[mt][nt][r] + bv;
        if (MODE == 1) {
          outF[(size_t)gm * Otot + go] = v;
        } else {
          int which = go >> 10, within = go & 1023;
          int h = within >> 6, d = within & 63;
          int b = gm >> 11, n = gm & 2047;
          size_t idx = (((size_t)b * 16 + h) * 2048 + n) * 64 + d;
          if (which == 0)      Qo[idx] = (__bf16)(v * ATTN_SCALE);
          else if (which == 1) Ko[idx] = (__bf16)v;
          else                 Vo[idx] = (__bf16)v;
        }
      }
    }
  }
}

// ---------------------------------------------------------------------------
// Flash attention over [B,H,N,64] bf16 Q(pre-scaled)/K/V.
// Block = 8 waves x 16 query rows = 128 rows; key tiles of 32.
// K tile staged by TDM (padded rows); V transposed into LDS by the waves.
// Output O: bf16 [B, N, 1024] (head-merged) for the projection GEMM.
// ---------------------------------------------------------------------------
__global__ __launch_bounds__(256)
void attn_kernel(const __bf16* __restrict__ Q, const __bf16* __restrict__ K,
                 const __bf16* __restrict__ V, __bf16* __restrict__ O) {
  __shared__ __align__(16) __bf16 Ks[32][80];      // [key][d], 160B stride
  __shared__ __align__(16) __bf16 Vt[64][40];      // [d][key], 80B stride
  __shared__ __align__(16) __bf16 Ps[8][16][40];   // per-wave P tile [m][key]

  const int tid  = threadIdx.x;
  const int lane = tid & 31, wave = tid >> 5;
  const int half = lane >> 4, l16 = lane & 15;
  const int bh = blockIdx.y;
  const int b = bh >> 4, h = bh & 15;
  const int q0 = blockIdx.x * 128 + wave * 16;

  const __bf16* Qb = Q + (size_t)bh * 2048 * 64;
  const __bf16* Kb = K + (size_t)bh * 2048 * 64;
  const u32* Vg = (const u32*)(V + (size_t)bh * 2048 * 64);

  // Q fragments: A 16x32, m = l16, elem e -> d = s*32 + (e&7) + ((e>>3)<<4) + (half<<3)
  v16bf qf[2];
#pragma unroll
  for (int s = 0; s < 2; ++s) {
    const __bf16* qrow = Qb + (size_t)(q0 + l16) * 64 + s * 32 + half * 8;
    qf[s] = cat8(*(const v8bf*)qrow, *(const v8bf*)(qrow + 16));
  }

  float mrow[8], lrow[8];
  v8f oacc[4];
#pragma unroll
  for (int r = 0; r < 8; ++r) { mrow[r] = -3.0e38f; lrow[r] = 0.0f; }
#pragma unroll
  for (int dt = 0; dt < 4; ++dt) oacc[dt] = vzero();

  for (int kb = 0; kb < 2048; kb += 32) {
    __syncthreads();
#if HAVE_TDM
    // K tile: 32 keys x 64 d; LDS rows 32 DW data + 8 DW pad -> 80-elem stride
    if (wave == 0) {
      tdm_load_2d((u32)(uintptr_t)&Ks[0][0], Kb + (size_t)kb * 64,
                  64, 64, 32, /*pad_i=*/4, /*pad_a=*/7, true);
    }
#endif
#pragma unroll
    for (int i = 0; i < 4; ++i) {
      int idx = tid + i * 256;          // 0..1023 -> 32 keys x 32 u32
      int r = idx >> 5, c = idx & 31;
#if !HAVE_TDM
      const u32* Kg = (const u32*)Kb;
      *(u32*)&Ks[r][c * 2] = Kg[(kb + r) * 32 + c];
#endif
      union { u32 u; __bf16 hh[2]; } cv;
      cv.u = Vg[(kb + r) * 32 + c];
      Vt[c * 2][r]     = cv.hh[0];      // transpose V into [d][key]
      Vt[c * 2 + 1][r] = cv.hh[1];
    }
#if HAVE_TDM
    if (wave == 0) __builtin_amdgcn_s_wait_tensorcnt(0);
#endif
    __syncthreads();

    // S = Q * K^T : two 16x16 score tiles (keys 0-15, 16-31)
    v8f s0 = vzero(), s1 = vzero();
#pragma unroll
    for (int s = 0; s < 2; ++s) {
      v16bf kf0 = cat8(*(const v8bf*)&Ks[l16][s * 32 + half * 8],
                       *(const v8bf*)&Ks[l16][s * 32 + half * 8 + 16]);
      v16bf kf1 = cat8(*(const v8bf*)&Ks[16 + l16][s * 32 + half * 8],
                       *(const v8bf*)&Ks[16 + l16][s * 32 + half * 8 + 16]);
      s0 = WMMA_BF16(qf[s], kf0, s0);
      s1 = WMMA_BF16(qf[s], kf1, s1);
    }

    // Online softmax per row (row m = r + 8*half, cols striped over 16 lanes)
#pragma unroll
    for (int r = 0; r < 8; ++r) {
      float rm = swz_max16(fmaxf(s0[r], s1[r]));
      float mnew = fmaxf(mrow[r], rm);
      float corr = __expf(mrow[r] - mnew);
      mrow[r] = mnew;
      float p0 = __expf(s0[r] - mnew);
      float p1 = __expf(s1[r] - mnew);
      float rs = swz_add16(p0 + p1);
      lrow[r] = lrow[r] * corr + rs;
#pragma unroll
      for (int dt = 0; dt < 4; ++dt) oacc[dt][r] *= corr;
      Ps[wave][half * 8 + r][l16]      = (__bf16)p0;
      Ps[wave][half * 8 + r][16 + l16] = (__bf16)p1;
    }

    // P (16x32) x V (32x64): A frag from Ps, B frags from transposed Vt
    v16bf pf = cat8(*(const v8bf*)&Ps[wave][l16][half * 8],
                    *(const v8bf*)&Ps[wave][l16][half * 8 + 16]);
#pragma unroll
    for (int dt = 0; dt < 4; ++dt) {
      v16bf vf = cat8(*(const v8bf*)&Vt[dt * 16 + l16][half * 8],
                      *(const v8bf*)&Vt[dt * 16 + l16][half * 8 + 16]);
      oacc[dt] = WMMA_BF16(pf, vf, oacc[dt]);
    }
  }

  // Normalize and write head-merged output [B, N, 1024]
#pragma unroll
  for (int dt = 0; dt < 4; ++dt) {
#pragma unroll
    for (int r = 0; r < 8; ++r) {
      int n = q0 + half * 8 + r;
      int d = dt * 16 + l16;
      float val = oacc[dt][r] / lrow[r];
      O[((size_t)b * 2048 + n) * 1024 + h * 64 + d] = (__bf16)val;
    }
  }
}

// ---------------------------------------------------------------------------
// Launch: cast -> QKV GEMM (scatter) -> flash attention -> projection GEMM
// ---------------------------------------------------------------------------
extern "C" void kernel_launch(void* const* d_in, const int* in_sizes, int n_in,
                              void* d_out, int out_size, void* d_ws, size_t ws_size,
                              hipStream_t stream) {
  const float* x      = (const float*)d_in[0];   // [2,2048,1024]
  const float* qkv_w  = (const float*)d_in[1];   // [3072,1024]
  const float* qkv_b  = (const float*)d_in[2];   // [3072]
  const float* proj_w = (const float*)d_in[3];   // [1024,1024]
  const float* proj_b = (const float*)d_in[4];   // [1024]
  float* out = (float*)d_out;                    // [2,2048,1024] f32

  char* ws = (char*)d_ws;
  __bf16* xb    = (__bf16*)(ws);                         //  8 MiB
  __bf16* wqkv  = (__bf16*)(ws + (size_t)8  * 1048576);  //  6 MiB
  __bf16* wproj = (__bf16*)(ws + (size_t)14 * 1048576);  //  2 MiB
  __bf16* Qb    = (__bf16*)(ws + (size_t)16 * 1048576);  //  8 MiB
  __bf16* Kb    = (__bf16*)(ws + (size_t)24 * 1048576);  //  8 MiB
  __bf16* Vb    = (__bf16*)(ws + (size_t)32 * 1048576);  //  8 MiB
  __bf16* Ob    = (__bf16*)(ws + (size_t)40 * 1048576);  //  8 MiB  (48 MiB total)

  cast_bf16_kernel<<<4096, 256, 0, stream>>>(x,      xb,    1048576);
  cast_bf16_kernel<<<3072, 256, 0, stream>>>(qkv_w,  wqkv,   786432);
  cast_bf16_kernel<<<1024, 256, 0, stream>>>(proj_w, wproj,  262144);

  // QKV: M=4096, O=3072  -> grid (O/128, M/128)
  gemm_bias_kernel<0><<<dim3(24, 32), 256, 0, stream>>>(
      xb, wqkv, qkv_b, nullptr, Qb, Kb, Vb, 3072);

  // attention: grid (N/128, B*H)
  attn_kernel<<<dim3(16, 32), 256, 0, stream>>>(Qb, Kb, Vb, Ob);

  // projection: M=4096, O=1024 -> f32 out
  gemm_bias_kernel<1><<<dim3(8, 32), 256, 0, stream>>>(
      Ob, wproj, proj_b, out, nullptr, nullptr, nullptr, 1024);
}
